// DynamicTransformer_76373108457724
// MI455X (gfx1250) — matrix operand order repaired and checked by reading, not compiled, not run on hardware
//
#include <hip/hip_runtime.h>
#include <hip/hip_bf16.h>

// ---------------- model dims (match reference) ----------------
constexpr int Vv  = 50257;
constexpr int Dm  = 512;
constexpr int NHn = 8;
constexpr int DFFn = 2048;
constexpr int Lx  = 8;
constexpr int Kr  = 4;
constexpr int Ss  = 512;
constexpr int Bb  = 4;
constexpr int RHn = 256;
constexpr int DHn = Dm / NHn;      // 64
constexpr float EPSc = 1e-5f;

typedef __attribute__((ext_vector_type(16))) _Float16 v16h;
typedef __attribute__((ext_vector_type(4)))  _Float16 v4h;
typedef __attribute__((ext_vector_type(8)))  float    v8f;

union FragH { v16h v; _Float16 e[16]; };
union FragF { v8f  v; float    e[8]; };

#define DEVI __device__ __forceinline__

// ---- fragment loaders from LDS (contiguous k order, same for A and B, wave32) ----
// A tile in LDS: row-major [M][ldA] halves, ldA multiple of 8 (16B-aligned rows).
// A frag (16x32 f16): lanes 0-15 -> M=0..15, K=0..15; lanes 16-31 -> K=16..31.
DEVI v16h frag_a(const _Float16* s, int ld, int row0, int k0, int lane) {
  FragH f;
  const int m  = lane & 15;
  const int kb = (lane >> 4) << 4;             // 0 or 16
  const _Float16* p = s + (size_t)(row0 + m) * ld + k0 + kb;
#pragma unroll
  for (int i = 0; i < 16; ++i) f.e[i] = p[i];        // contiguous: two b128
  return f.v;
}
// B tile stored TRANSPOSED in LDS: [N][ldB] halves (k contiguous per column).
// B frag (32x16 f16): lanes 0-15 -> N=0..15, K=0..15; lanes 16-31 -> K=16..31.
DEVI v16h frag_bt(const _Float16* s, int ld, int k0, int col0, int lane) {
  FragH f;
  const int n  = lane & 15;
  const int kb = (lane >> 4) << 4;             // 0 or 16
  const _Float16* p = s + (size_t)(col0 + n) * ld + k0 + kb;
#pragma unroll
  for (int i = 0; i < 16; ++i) f.e[i] = p[i];        // contiguous: two b128
  return f.v;
}

DEVI v8f v8f_zero() {
  v8f z = {0.f, 0.f, 0.f, 0.f, 0.f, 0.f, 0.f, 0.f};
  return z;
}
DEVI void cvt_store4(_Float16* dst, float4 f) {      // ds_store_b64 of 4 cvt'd halves
  v4h h = {(_Float16)f.x, (_Float16)f.y, (_Float16)f.z, (_Float16)f.w};
  *(v4h*)dst = h;
}

// ---------------- embedding ----------------
__global__ __launch_bounds__(256) void embed_kernel(
    const int* __restrict__ tokens, const float* __restrict__ emb,
    const float* __restrict__ pos, float* __restrict__ X) {
  const int row = blockIdx.x;              // b*Ss + s
  const int s   = row % Ss;
  const int tok = tokens[row];
  const int t   = threadIdx.x;
  X[(size_t)row * Dm + t]       = emb[(size_t)tok * Dm + t]       + pos[(size_t)s * Dm + t];
  X[(size_t)row * Dm + t + 256] = emb[(size_t)tok * Dm + t + 256] + pos[(size_t)s * Dm + t + 256];
}

// ---------------- mean pool over S ----------------
__global__ __launch_bounds__(512) void pool_kernel(
    const float* __restrict__ X, float* __restrict__ pooled) {
  const int b = blockIdx.x, d = threadIdx.x;
  float acc = 0.f;
  for (int si = 0; si < Ss; ++si) acc += X[((size_t)b * Ss + si) * Dm + d];
  pooled[(size_t)b * Dm + d] = acc * (1.f / Ss);
}

// ---------------- router MLP + argmax ----------------
__global__ __launch_bounds__(RHn) void router_kernel(
    const float* __restrict__ pooled,
    const float* __restrict__ w1, const float* __restrict__ b1,
    const float* __restrict__ w2, const float* __restrict__ b2,
    float* __restrict__ rlog_out, int* __restrict__ choice_out,
    int* __restrict__ idx_ws, int round) {
  const int b = blockIdx.x, r = threadIdx.x;
  __shared__ float hsh[RHn];
  __shared__ float rls[Lx];
  const float* pb = pooled + (size_t)b * Dm;
  float a = b1[r];
  for (int d = 0; d < Dm; ++d) a += pb[d] * w1[(size_t)d * RHn + r];
  hsh[r] = fmaxf(a, 0.f);
  __syncthreads();
  if (r < Lx) {
    float a2 = b2[r];
    for (int j = 0; j < RHn; ++j) a2 += hsh[j] * w2[(size_t)j * Lx + r];
    rls[r] = a2;                                      // TEMP == 1.0
    rlog_out[((size_t)b * Kr + round) * Lx + r] = a2;
  }
  __syncthreads();
  if (r == 0) {
    int best = 0; float bv = rls[0];
    for (int l = 1; l < Lx; ++l) if (rls[l] > bv) { bv = rls[l]; best = l; }
    choice_out[b * Kr + round] = best;
    idx_ws[b] = best;
  }
}

// ---------------- layernorm (routed scale/bias pools) ----------------
__global__ __launch_bounds__(256) void ln_kernel(
    const float* __restrict__ X, const float* __restrict__ sPool,
    const float* __restrict__ bPool, const int* __restrict__ idx,
    float* __restrict__ H) {
  const int row = blockIdx.x;               // b*Ss + s
  const int b   = row / Ss;
  const int l   = idx[b];
  const int t   = threadIdx.x;
  const float* x = X + (size_t)row * Dm;
  __shared__ float red[256];
  const float v0 = x[t], v1 = x[t + 256];
  red[t] = v0 + v1; __syncthreads();
  for (int s2 = 128; s2 > 0; s2 >>= 1) { if (t < s2) red[t] += red[t + s2]; __syncthreads(); }
  const float mean = red[0] * (1.f / Dm);
  __syncthreads();
  const float d0 = v0 - mean, d1 = v1 - mean;
  red[t] = d0 * d0 + d1 * d1; __syncthreads();
  for (int s2 = 128; s2 > 0; s2 >>= 1) { if (t < s2) red[t] += red[t + s2]; __syncthreads(); }
  const float inv = rsqrtf(red[0] * (1.f / Dm) + EPSc);
  const float* sc = sPool + (size_t)l * Dm;
  const float* bi = bPool + (size_t)l * Dm;
  H[(size_t)row * Dm + t]       = d0 * inv * sc[t]       + bi[t];
  H[(size_t)row * Dm + t + 256] = d1 * inv * sc[t + 256] + bi[t + 256];
}

// ---------------- generic routed WMMA GEMM ----------------
// C[b] = A[b] @ Wpool[idx[b]] (+bias) (+resid) (relu?), f16 compute f32 accum.
// Block tile 128x128, BK=32; 8 waves (4x2), each wave 32x64 (2x4 WMMA tiles).
constexpr int LDA = 40;   // 32 + 8 pad, 80B rows (16B aligned)
constexpr int LDB = 40;
__global__ __launch_bounds__(256) void gemm_wmma_kernel(
    const float* __restrict__ A, const float* __restrict__ Wpool,
    const float* __restrict__ biasPool, const int* __restrict__ idx,
    const float* resid, float* C,
    int Sm, int Kd, int Nn, int applyRelu) {
  __shared__ __align__(16) _Float16 As[128 * LDA];   // [M][K] row-major
  __shared__ __align__(16) _Float16 Bst[128 * LDB];  // [N][K] transposed
  const int tid  = threadIdx.x;
  const int lane = tid & 31;
  const int wave = tid >> 5;
  const int wm   = wave & 3, wn = wave >> 2;        // wave tile origin (wm*32, wn*64)
  const int b    = blockIdx.z;
  const int row0 = blockIdx.y * 128;
  const int col0 = blockIdx.x * 128;
  const int layer = idx ? idx[b] : 0;
  const float* Ab   = A + (size_t)b * Sm * Kd;
  const float* W    = Wpool + (size_t)layer * Kd * Nn;
  const float* bias = biasPool ? (biasPool + (size_t)layer * Nn) : nullptr;
  const float* Rb   = resid ? (resid + (size_t)b * Sm * Nn) : nullptr;
  float*       Cb   = C + (size_t)b * Sm * Nn;
  // uniform per block: fast (guard-free, vectorizable) B path?
  const bool fastB = ((Nn & 3) == 0) && (col0 + 128 <= Nn);

  v8f acc[2][4];
#pragma unroll
  for (int i = 0; i < 2; ++i)
#pragma unroll
    for (int j = 0; j < 4; ++j) acc[i][j] = v8f_zero();

  for (int k0 = 0; k0 < Kd; k0 += 32) {
    // ---- A tile: float4 global loads (rows 16B aligned: Kd % 32 == 0), b64 LDS stores
#pragma unroll
    for (int p = 0; p < 4; ++p) {
      const int e = (tid + p * 256) * 4;            // 0..4095
      const int r = e >> 5, c = e & 31;             // c multiple of 4
      const float4 f = *(const float4*)&Ab[(size_t)(row0 + r) * Kd + k0 + c];
      cvt_store4(&As[r * LDA + c], f);
    }
    // ---- B tile: coalesced global read, transposed LDS store
    if (fastB) {
#pragma unroll
      for (int p = 0; p < 4; ++p) {
        const int e = (tid + p * 256) * 4;
        const int c = e & 127, r = e >> 7;          // c = n-within-tile (mult of 4), r = k
        const float4 f = *(const float4*)&W[(size_t)(k0 + r) * Nn + col0 + c];
        Bst[(c + 0) * LDB + r] = (_Float16)f.x;
        Bst[(c + 1) * LDB + r] = (_Float16)f.y;
        Bst[(c + 2) * LDB + r] = (_Float16)f.z;
        Bst[(c + 3) * LDB + r] = (_Float16)f.w;
      }
    } else {
#pragma unroll
      for (int p = 0; p < 16; ++p) {
        const int t = tid + p * 256;
        const int c = t & 127, r = t >> 7;
        const int n = col0 + c;
        Bst[c * LDB + r] = (n < Nn) ? (_Float16)W[(size_t)(k0 + r) * Nn + n]
                                    : (_Float16)0.f;
      }
    }
    __syncthreads();
    v16h af[2], bf[4];
#pragma unroll
    for (int i = 0; i < 2; ++i) af[i] = frag_a(As, LDA, wm * 32 + i * 16, 0, lane);
#pragma unroll
    for (int j = 0; j < 4; ++j) bf[j] = frag_bt(Bst, LDB, 0, wn * 64 + j * 16, lane);
#pragma unroll
    for (int i = 0; i < 2; ++i)
#pragma unroll
      for (int j = 0; j < 4; ++j)
        acc[i][j] = __builtin_amdgcn_wmma_f32_16x16x32_f16(
            false, af[i], false, bf[j], (short)0, acc[i][j], false, false);
    __syncthreads();
  }

  const int nlo = lane & 15;
  const int mo  = (lane >> 4) * 8;
#pragma unroll
  for (int i = 0; i < 2; ++i)
#pragma unroll
    for (int j = 0; j < 4; ++j) {
      FragF fc; fc.v = acc[i][j];
      const int n = col0 + wn * 64 + j * 16 + nlo;
      if (fastB || n < Nn) {
        const float bv = bias ? bias[n] : 0.f;
#pragma unroll
        for (int r = 0; r < 8; ++r) {
          const int m = row0 + wm * 32 + i * 16 + mo + r;
          float val = fc.e[r] + bv;
          if (Rb) val += Rb[(size_t)m * Nn + n];
          if (applyRelu) val = fmaxf(val, 0.f);
          Cb[(size_t)m * Nn + n] = val;
        }
      }
    }
}

// ---------------- attention scores: S[q,k] = (Q·K^T)/8, per (b,h) ----------------
constexpr int LDQ = 72;   // 64 + 8 pad (144B rows, 16B aligned)
__global__ __launch_bounds__(256) void attn_scores_kernel(
    const float* __restrict__ Q, const float* __restrict__ Kt,
    float* __restrict__ scores) {
  __shared__ __align__(16) _Float16 As[128 * LDQ];  // Q tile [q][d]
  __shared__ __align__(16) _Float16 Bst[64 * LDQ];  // K tile [k_col][d] (natural layout)
  const int tid  = threadIdx.x, lane = tid & 31, wave = tid >> 5;
  const int wm   = wave & 3, wn = wave >> 2;
  const int bh   = blockIdx.z;
  const int b    = bh / NHn, h = bh % NHn;
  const int q0   = blockIdx.y * 128;
  const int n0   = blockIdx.x * 64;
  const float* Qb = Q  + (size_t)b * Ss * Dm + h * DHn;
  const float* Kb = Kt + (size_t)b * Ss * Dm + h * DHn;

#pragma unroll
  for (int p = 0; p < 8; ++p) {                     // 128x64 A tile, float4
    const int e = (tid + p * 256) * 4;
    const int r = e >> 6, c = e & 63;
    cvt_store4(&As[r * LDQ + c], *(const float4*)&Qb[(size_t)(q0 + r) * Dm + c]);
  }
#pragma unroll
  for (int p = 0; p < 4; ++p) {                     // 64x64 B tile, float4, no transpose
    const int e = (tid + p * 256) * 4;
    const int r = e >> 6, c = e & 63;
    cvt_store4(&Bst[r * LDQ + c], *(const float4*)&Kb[(size_t)(n0 + r) * Dm + c]);
  }
  __syncthreads();

  v8f acc[2][2];
#pragma unroll
  for (int i = 0; i < 2; ++i)
#pragma unroll
    for (int j = 0; j < 2; ++j) acc[i][j] = v8f_zero();

#pragma unroll
  for (int k0 = 0; k0 < 64; k0 += 32) {
    v16h af[2], bf[2];
#pragma unroll
    for (int i = 0; i < 2; ++i) af[i] = frag_a(As, LDQ, wm * 32 + i * 16, k0, lane);
#pragma unroll
    for (int j = 0; j < 2; ++j) bf[j] = frag_bt(Bst, LDQ, k0, wn * 32 + j * 16, lane);
#pragma unroll
    for (int i = 0; i < 2; ++i)
#pragma unroll
      for (int j = 0; j < 2; ++j)
        acc[i][j] = __builtin_amdgcn_wmma_f32_16x16x32_f16(
            false, af[i], false, bf[j], (short)0, acc[i][j], false, false);
  }

  float* Sb = scores + (size_t)bh * Ss * Ss;
  const int nlo = lane & 15, mo = (lane >> 4) * 8;
#pragma unroll
  for (int i = 0; i < 2; ++i)
#pragma unroll
    for (int j = 0; j < 2; ++j) {
      FragF fc; fc.v = acc[i][j];
      const int n = n0 + wn * 32 + j * 16 + nlo;
#pragma unroll
      for (int r = 0; r < 8; ++r) {
        const int m = q0 + wm * 32 + i * 16 + mo + r;
        Sb[(size_t)m * Ss + n] = fc.e[r] * 0.125f;   // 1/sqrt(64)
      }
    }
}

// ---------------- row softmax over k (rows = B*NH*S, len 512) ----------------
__global__ __launch_bounds__(256) void softmax_kernel(float* __restrict__ scores) {
  float* p = scores + (size_t)blockIdx.x * Ss;
  const int t = threadIdx.x;
  const float a = p[t], c = p[t + 256];
  __shared__ float red[256];
  red[t] = fmaxf(a, c); __syncthreads();
  for (int s2 = 128; s2 > 0; s2 >>= 1) { if (t < s2) red[t] = fmaxf(red[t], red[t + s2]); __syncthreads(); }
  const float mx = red[0]; __syncthreads();
  const float e0 = __expf(a - mx), e1 = __expf(c - mx);
  red[t] = e0 + e1; __syncthreads();
  for (int s2 = 128; s2 > 0; s2 >>= 1) { if (t < s2) red[t] += red[t + s2]; __syncthreads(); }
  const float inv = 1.f / red[0];
  p[t] = e0 * inv; p[t + 256] = e1 * inv;
}

// ---------------- O_head = attn @ V_head (M=S, N=64, K=S) ----------------
__global__ __launch_bounds__(256) void attn_av_kernel(
    const float* __restrict__ P, const float* __restrict__ Vm,
    float* __restrict__ O) {
  __shared__ __align__(16) _Float16 As[128 * LDA];  // probs tile [q][k]
  __shared__ __align__(16) _Float16 Bst[64 * LDB];  // V tile [n(d)][k] transposed
  const int tid  = threadIdx.x, lane = tid & 31, wave = tid >> 5;
  const int wm   = wave & 3, wn = wave >> 2;
  const int bh   = blockIdx.z;
  const int b    = bh / NHn, h = bh % NHn;
  const int row0 = blockIdx.y * 128;
  const float* Ab = P  + (size_t)bh * Ss * Ss;
  const float* Vb = Vm + (size_t)b * Ss * Dm + h * DHn;

  v8f acc[2][2];
#pragma unroll
  for (int i = 0; i < 2; ++i)
#pragma unroll
    for (int j = 0; j < 2; ++j) acc[i][j] = v8f_zero();

  for (int k0 = 0; k0 < Ss; k0 += 32) {
#pragma unroll
    for (int p = 0; p < 4; ++p) {                   // 128x32 probs tile, float4
      const int e = (tid + p * 256) * 4;
      const int r = e >> 5, c = e & 31;
      cvt_store4(&As[r * LDA + c], *(const float4*)&Ab[(size_t)(row0 + r) * Ss + k0 + c]);
    }
#pragma unroll
    for (int p = 0; p < 2; ++p) {                   // 32x64 V tile, float4 + transposed store
      const int e = (tid + p * 256) * 4;
      const int c = e & 63, r = e >> 6;             // c = d (mult of 4), r = k
      const float4 f = *(const float4*)&Vb[(size_t)(k0 + r) * Dm + c];
      Bst[(c + 0) * LDB + r] = (_Float16)f.x;
      Bst[(c + 1) * LDB + r] = (_Float16)f.y;
      Bst[(c + 2) * LDB + r] = (_Float16)f.z;
      Bst[(c + 3) * LDB + r] = (_Float16)f.w;
    }
    __syncthreads();
    v16h af[2], bf[2];
#pragma unroll
    for (int i = 0; i < 2; ++i) af[i] = frag_a(As, LDA, wm * 32 + i * 16, 0, lane);
#pragma unroll
    for (int j = 0; j < 2; ++j) bf[j] = frag_bt(Bst, LDB, 0, wn * 32 + j * 16, lane);
#pragma unroll
    for (int i = 0; i < 2; ++i)
#pragma unroll
      for (int j = 0; j < 2; ++j)
        acc[i][j] = __builtin_amdgcn_wmma_f32_16x16x32_f16(
            false, af[i], false, bf[j], (short)0, acc[i][j], false, false);
    __syncthreads();
  }

  const int nlo = lane & 15, mo = (lane >> 4) * 8;
#pragma unroll
  for (int i = 0; i < 2; ++i)
#pragma unroll
    for (int j = 0; j < 2; ++j) {
      FragF fc; fc.v = acc[i][j];
      const int n = wn * 32 + j * 16 + nlo;     // 0..63 within head
#pragma unroll
      for (int r = 0; r < 8; ++r) {
        const int m = row0 + wm * 32 + i * 16 + mo + r;
        O[((size_t)b * Ss + m) * Dm + h * DHn + n] = fc.e[r];
      }
    }
}

// ---------------- host orchestration ----------------
extern "C" void kernel_launch(void* const* d_in, const int* in_sizes, int n_in,
                              void* d_out, int out_size, void* d_ws, size_t ws_size,
                              hipStream_t stream) {
  (void)in_sizes; (void)n_in; (void)out_size; (void)ws_size;
  const int*   tokens = (const int*)d_in[0];
  const float* emb  = (const float*)d_in[1];
  const float* pos  = (const float*)d_in[2];
  const float* rw1  = (const float*)d_in[3];
  const float* rb1  = (const float*)d_in[4];
  const float* rw2  = (const float*)d_in[5];
  const float* rb2  = (const float*)d_in[6];
  const float* ln1s = (const float*)d_in[7];
  const float* ln1b = (const float*)d_in[8];
  const float* wq   = (const float*)d_in[9];
  const float* bq   = (const float*)d_in[10];
  const float* wk   = (const float*)d_in[11];
  const float* bk   = (const float*)d_in[12];
  const float* wv   = (const float*)d_in[13];
  const float* bvv  = (const float*)d_in[14];
  const float* wo   = (const float*)d_in[15];
  const float* bo   = (const float*)d_in[16];
  const float* ln2s = (const float*)d_in[17];
  const float* ln2b = (const float*)d_in[18];
  const float* wf1  = (const float*)d_in[19];
  const float* bf1  = (const float*)d_in[20];
  const float* wf2  = (const float*)d_in[21];
  const float* bf2  = (const float*)d_in[22];
  const float* outw = (const float*)d_in[23];
  const float* outb = (const float*)d_in[24];

  // outputs, packed in reference return order
  float* logits  = (float*)d_out;                                  // [B,S,V] f32
  int*   choices = (int*)(logits + (size_t)Bb * Ss * Vv);          // [B,K]   i32
  float* rlog    = (float*)(choices + Bb * Kr);                    // [B,K,L] f32

  // workspace carve-up (floats)
  float* ws = (float*)d_ws;
  size_t off = 0;
  const size_t nXD = (size_t)Bb * Ss * Dm;
  float* X    = ws + off; off += nXD;                              // activations
  float* H    = ws + off; off += nXD;                              // post-LN
  float* Qb   = ws + off; off += nXD;
  float* Kb   = ws + off; off += nXD;
  float* Vb   = ws + off; off += nXD;
  float* Ob   = ws + off; off += nXD;
  float* SC   = ws + off; off += (size_t)Bb * NHn * Ss * Ss;       // scores/probs
  float* FF   = ws + off; off += (size_t)Bb * Ss * DFFn;           // FFN hidden
  float* pooled = ws + off; off += (size_t)Bb * Dm;
  int* idxb = (int*)(ws + off); off += 16;                         // routed layer per b

  embed_kernel<<<Bb * Ss, 256, 0, stream>>>(tokens, emb, pos, X);

  const dim3 gD (Dm   / 128, Ss / 128, Bb);        // [S,D] routed GEMMs
  const dim3 gF1(DFFn / 128, Ss / 128, Bb);        // FFN up
  const dim3 gSc(Ss / 64,   Ss / 128, Bb * NHn);   // attention scores
  const dim3 gAv(1,         Ss / 128, Bb * NHn);   // attn @ V

  for (int round = 0; round < Kr; ++round) {
    pool_kernel  <<<Bb, Dm, 0, stream>>>(X, pooled);
    router_kernel<<<Bb, RHn, 0, stream>>>(pooled, rw1, rb1, rw2, rb2,
                                          rlog, choices, idxb, round);
    ln_kernel<<<Bb * Ss, 256, 0, stream>>>(X, ln1s, ln1b, idxb, H);
    gemm_wmma_kernel<<<gD, 256, 0, stream>>>(H, wq, bq, idxb, nullptr, Qb, Ss, Dm, Dm, 0);
    gemm_wmma_kernel<<<gD, 256, 0, stream>>>(H, wk, bk, idxb, nullptr, Kb, Ss, Dm, Dm, 0);
    gemm_wmma_kernel<<<gD, 256, 0, stream>>>(H, wv, bvv, idxb, nullptr, Vb, Ss, Dm, Dm, 0);
    attn_scores_kernel<<<gSc, 256, 0, stream>>>(Qb, Kb, SC);
    softmax_kernel<<<Bb * NHn * Ss, 256, 0, stream>>>(SC);
    attn_av_kernel<<<gAv, 256, 0, stream>>>(SC, Vb, Ob);
    gemm_wmma_kernel<<<gD, 256, 0, stream>>>(Ob, wo, bo, idxb, X, X, Ss, Dm, Dm, 0);
    ln_kernel<<<Bb * Ss, 256, 0, stream>>>(X, ln2s, ln2b, idxb, H);
    gemm_wmma_kernel<<<gF1, 256, 0, stream>>>(H, wf1, bf1, idxb, nullptr, FF, Ss, Dm, DFFn, 1);
    gemm_wmma_kernel<<<gD, 256, 0, stream>>>(FF, wf2, bf2, idxb, X, X, Ss, DFFn, Dm, 0);
  }

  // final vocab projection: [B*S, D] @ [D, V] + out_b  (N-tail guarded)
  const dim3 gV((Vv + 127) / 128, (Bb * Ss) / 128, 1);
  gemm_wmma_kernel<<<gV, 256, 0, stream>>>(X, outw, outb, nullptr, nullptr,
                                           logits, Bb * Ss, Dm, Vv, 0);
}
